// IRLoss_78159814852941
// MI455X (gfx1250) — matrix operand order repaired and checked by reading, not compiled, not run on hardware
//
#include <hip/hip_runtime.h>

// ---------------------------------------------------------------------------
// MI455X (gfx1250) fused cosine-similarity overlap loss.
//
// Roofline: ~34 GFLOP of batched GEMM (N=32, M=N=1024, K=256) vs 134 MB of
// input (~6 us @ 23.3 TB/s) -> matrix-compute bound; entire working set fits
// in the 192 MB L2. Strategy: L2-normalize once, split fp32 into bf16 hi/lo,
// run 3x V_WMMA_F32_16X16X32_BF16 per tile (error-compensated bf16x3, ~fp32
// accuracy), and fuse the geometric pos_mask + reduction into the GEMM
// epilogue so the 1024x1024 cosine matrix never touches memory.
//
// GEMM blocking: 256 threads = 8 waves; block tile 128x128; wave tile 32x64
// (2 M-tiles x 4 N-tiles -> 24 WMMAs per 24 ds_load_b128 per K-chunk).
// Next K-chunk is prefetched into registers while the current chunk computes.
// ---------------------------------------------------------------------------

typedef __attribute__((ext_vector_type(16))) __bf16 v16bf;
typedef __attribute__((ext_vector_type(8)))  __bf16 v8bf;
typedef __attribute__((ext_vector_type(8)))  float  v8f;

#define HW   1024       // H*W = 32*32
#define CC   256        // channels (GEMM K)
#define LDSR 40         // LDS row stride in bf16 elems (80 B: 16B-aligned, conflict-free)

__device__ __forceinline__ v16bf cat8(v8bf a, v8bf b) {
  return __builtin_shufflevector(a, b, 0,1,2,3,4,5,6,7,8,9,10,11,12,13,14,15);
}

// ---------------- Phase A: per-position L2 normalize + bf16 hi/lo split -----
// in : [N][C][HW] fp32   out: [N][HW][C] bf16 (hi plane, lo plane) transposed
__global__ void normalize_split_kernel(const float* __restrict__ in,
                                       __bf16* __restrict__ out_hi,
                                       __bf16* __restrict__ out_lo) {
  __shared__ float tile[CC * 33];   // [c][p] padded: bank = (33c+p)%64, conflict-free
  __shared__ float partial[8 * 32];
  __shared__ float rnorm[32];
  const int n   = blockIdx.y;
  const int p0  = blockIdx.x * 32;
  const int tid = threadIdx.x;          // 256 threads
  const int pl  = tid & 31;
  const int cg  = tid >> 5;             // 8 channel groups of 32

  const float* src = in + (size_t)n * CC * HW + p0 + pl;
  float acc = 0.f;
#pragma unroll 4
  for (int k = 0; k < 32; ++k) {
    int c = cg * 32 + k;
    float v = src[(size_t)c * HW];      // coalesced over pl
    tile[c * 33 + pl] = v;
    acc += v * v;
  }
  partial[cg * 32 + pl] = acc;
  __syncthreads();
  if (tid < 32) {
    float s = 0.f;
#pragma unroll
    for (int g = 0; g < 8; ++g) s += partial[g * 32 + tid];
    rnorm[tid] = 1.0f / fmaxf(sqrtf(s), 1e-8f);
  }
  __syncthreads();

  // thread owns channel c = tid, loops the 32 positions -> coalesced bf16 writes
  __bf16* dh = out_hi + ((size_t)n * HW + p0) * CC + tid;
  __bf16* dl = out_lo + ((size_t)n * HW + p0) * CC + tid;
#pragma unroll 4
  for (int it = 0; it < 32; ++it) {
    float v  = tile[tid * 33 + it] * rnorm[it];
    __bf16 h = (__bf16)v;
    __bf16 l = (__bf16)(v - (float)h);
    dh[(size_t)it * CC] = h;
    dl[(size_t)it * CC] = l;
  }
}

// ---------------- Phase B: fused WMMA GEMM + pos_mask reduction -------------
// A = q-hat [N][HW][C], B = k-hat [N][HW][C] (both bf16 hi/lo planes).
// Block: 256 threads (8 waves), 128x128 tile; wave = 32x64 (waveM 0..3, waveN 0..1).
__global__ void __launch_bounds__(256)
cossim_wmma_kernel(const __bf16* __restrict__ Ahi, const __bf16* __restrict__ Alo,
                   const __bf16* __restrict__ Bhi, const __bf16* __restrict__ Blo,
                   const float* __restrict__ coord_q, const float* __restrict__ coord_k,
                   float* __restrict__ part_sum, float* __restrict__ part_cnt) {
  __shared__ __attribute__((aligned(16))) __bf16 lAh[128 * LDSR];
  __shared__ __attribute__((aligned(16))) __bf16 lAl[128 * LDSR];
  __shared__ __attribute__((aligned(16))) __bf16 lBh[128 * LDSR];
  __shared__ __attribute__((aligned(16))) __bf16 lBl[128 * LDSR];
  __shared__ float redS[256], redC[256];

  const int n     = blockIdx.z;
  const int pm0   = blockIdx.x * 128;    // output rows  (p)
  const int qn0   = blockIdx.y * 128;    // output cols  (q)
  const int tid   = threadIdx.x;
  const int lane  = tid & 31;
  const int wave  = tid >> 5;
  const int waveM = wave >> 1;           // 0..3 -> M offset 32*waveM
  const int waveN = wave & 1;            // 0..1 -> N offset 64*waveN

  const __bf16* srcs[4] = { Ahi + ((size_t)n * HW + pm0) * CC,
                            Alo + ((size_t)n * HW + pm0) * CC,
                            Bhi + ((size_t)n * HW + qn0) * CC,
                            Blo + ((size_t)n * HW + qn0) * CC };
  __bf16* dsts[4] = { lAh, lAl, lBh, lBl };

  v8f acc[2][4] = {};

  // per-lane fragment addressing (documented CDNA5 16-bit WMMA layouts)
  const int mrow = (lane & 15);
  const int kA   = (lane >> 4) << 3;     // A: lanes<16 -> K {0..7,16..23}; else +8
  const int kB   = (lane >> 4) << 3;     // B: lanes<16 -> K {0..7,16..23}; else +8

  const int lrow = tid >> 1;             // global->LDS staging map: 128 rows
  const int lseg = (tid & 1) * 16;       // bf16 element offset (16B aligned)

  // register double buffer for the next K-chunk
  v8bf st[4][2];
#pragma unroll
  for (int pl = 0; pl < 4; ++pl) {
    const __bf16* s = srcs[pl] + (size_t)lrow * CC + lseg;
    st[pl][0] = *(const v8bf*)(s);
    st[pl][1] = *(const v8bf*)(s + 8);
  }

  for (int kc = 0; kc < CC / 32; ++kc) {
    __syncthreads();                     // previous compute done -> LDS reusable
#pragma unroll
    for (int pl = 0; pl < 4; ++pl) {
      *(v8bf*)&dsts[pl][lrow * LDSR + lseg]     = st[pl][0];
      *(v8bf*)&dsts[pl][lrow * LDSR + lseg + 8] = st[pl][1];
    }
    __syncthreads();

    if (kc + 1 < CC / 32) {              // prefetch next chunk under the WMMAs
      const int kb = (kc + 1) * 32;
#pragma unroll
      for (int pl = 0; pl < 4; ++pl) {
        const __bf16* s = srcs[pl] + (size_t)lrow * CC + kb + lseg;
        st[pl][0] = *(const v8bf*)(s);
        st[pl][1] = *(const v8bf*)(s + 8);
      }
    }

    v16bf ahf[2], alf[2];
#pragma unroll
    for (int mi = 0; mi < 2; ++mi) {
      const int ar = (waveM * 32 + mi * 16 + mrow) * LDSR + kA;
      ahf[mi] = cat8(*(const v8bf*)&lAh[ar], *(const v8bf*)&lAh[ar + 16]);
      alf[mi] = cat8(*(const v8bf*)&lAl[ar], *(const v8bf*)&lAl[ar + 16]);
    }
#pragma unroll
    for (int j = 0; j < 4; ++j) {
      const int br = (waveN * 64 + j * 16 + mrow) * LDSR + kB;
      v16bf bh = cat8(*(const v8bf*)&lBh[br], *(const v8bf*)&lBh[br + 16]);
      v16bf bl = cat8(*(const v8bf*)&lBl[br], *(const v8bf*)&lBl[br + 16]);
#pragma unroll
      for (int mi = 0; mi < 2; ++mi) {
        // bf16x3: hi*hi + hi*lo + lo*hi accumulated in fp32
        acc[mi][j] = __builtin_amdgcn_wmma_f32_16x16x32_bf16(false, ahf[mi], false, bh, (short)0, acc[mi][j], false, false);
        acc[mi][j] = __builtin_amdgcn_wmma_f32_16x16x32_bf16(false, ahf[mi], false, bl, (short)0, acc[mi][j], false, false);
        acc[mi][j] = __builtin_amdgcn_wmma_f32_16x16x32_bf16(false, alf[mi], false, bh, (short)0, acc[mi][j], false, false);
      }
    }
  }

  // ---- epilogue: analytic pos_mask, fused reduction ----
  const float qx1 = coord_q[n * 4 + 0], qy1 = coord_q[n * 4 + 1];
  const float qx2 = coord_q[n * 4 + 2], qy2 = coord_q[n * 4 + 3];
  const float kx1 = coord_k[n * 4 + 0], ky1 = coord_k[n * 4 + 1];
  const float kx2 = coord_k[n * 4 + 2], ky2 = coord_k[n * 4 + 3];
  const float q_bw = (qx2 - qx1) * (1.0f / 32.0f), q_bh = (qy2 - qy1) * (1.0f / 32.0f);
  const float k_bw = (kx2 - kx1) * (1.0f / 32.0f), k_bh = (ky2 - ky1) * (1.0f / 32.0f);
  const float qd2 = q_bw * q_bw + q_bh * q_bh;
  const float kd2 = k_bw * k_bw + k_bh * k_bh;
  const float thr2 = 256.0f * fmaxf(qd2, kd2);   // (pos_ratio*H)^2 * max_diag^2

  float lsum = 0.f, lcnt = 0.f;
  const int phi = (lane >> 4) << 3;              // C layout: lanes>=16 -> M += 8
#pragma unroll
  for (int j = 0; j < 4; ++j) {
    const int qg = qn0 + waveN * 64 + j * 16 + (lane & 15);  // N = lane&15
    const float ckx = ((qg & 31) + 0.5f) * k_bw + kx1;
    const float cky = ((qg >> 5) + 0.5f) * k_bh + ky1;
#pragma unroll
    for (int mi = 0; mi < 2; ++mi) {
      const int pb = pm0 + waveM * 32 + mi * 16 + phi;
#pragma unroll
      for (int r = 0; r < 8; ++r) {
        const int pg = pb + r;                   // M = r (+8 for upper lanes)
        const float cqx = ((pg & 31) + 0.5f) * q_bw + qx1;
        const float cqy = ((pg >> 5) + 0.5f) * q_bh + qy1;
        const float dx = cqx - ckx, dy = cqy - cky;
        if (dx * dx + dy * dy < thr2) { lsum += acc[mi][j][r]; lcnt += 1.0f; }
      }
    }
  }

  redS[tid] = lsum; redC[tid] = lcnt;
  __syncthreads();
  for (int s = 128; s > 0; s >>= 1) {
    if (tid < s) { redS[tid] += redS[tid + s]; redC[tid] += redC[tid + s]; }
    __syncthreads();
  }
  if (tid == 0) {
    const int per = gridDim.x * gridDim.y;       // 64 blocks per batch
    const int bid = blockIdx.y * gridDim.x + blockIdx.x;
    part_sum[(size_t)n * per + bid] = redS[0];
    part_cnt[(size_t)n * per + bid] = redC[0];
  }
}

// ---------------- Phase C: deterministic final reduction --------------------
__global__ void finalize_kernel(const float* __restrict__ part_sum,
                                const float* __restrict__ part_cnt,
                                int pairs, int per, float* __restrict__ out) {
  __shared__ float s4[256], c4[256], ratio[64];
  const int tid = threadIdx.x;
  const int pair = tid >> 2, sub = tid & 3;
  float s = 0.f, c = 0.f;
  if (pair < pairs) {
    for (int i = sub; i < per; i += 4) {
      s += part_sum[(size_t)pair * per + i];
      c += part_cnt[(size_t)pair * per + i];
    }
  }
  s4[tid] = s; c4[tid] = c;
  __syncthreads();
  if (sub == 0 && pair < pairs) {
    float ts = s4[tid] + s4[tid + 1] + s4[tid + 2] + s4[tid + 3];
    float tc = c4[tid] + c4[tid + 1] + c4[tid + 2] + c4[tid + 3];
    ratio[pair] = ts / (tc + 1e-6f);
  }
  __syncthreads();
  if (tid == 0) {
    float t = 0.f;
    for (int i = 0; i < pairs; ++i) t += ratio[i];
    out[0] = -t / (float)pairs;   // == -(l1.mean()+l2.mean())*0.5
  }
}

// ---------------------------------------------------------------------------
extern "C" void kernel_launch(void* const* d_in, const int* in_sizes, int n_in,
                              void* d_out, int out_size, void* d_ws, size_t ws_size,
                              hipStream_t stream) {
  const float* p1 = (const float*)d_in[0];
  const float* p2 = (const float*)d_in[1];
  const float* z1 = (const float*)d_in[2];
  const float* z2 = (const float*)d_in[3];
  const float* c1 = (const float*)d_in[4];
  const float* c2 = (const float*)d_in[5];
  const int N = in_sizes[0] / (CC * HW);          // 32

  const size_t plane = (size_t)N * HW * CC;       // bf16 elems per plane
  __bf16* ws  = (__bf16*)d_ws;
  __bf16* p1h = ws + 0 * plane; __bf16* p1l = ws + 1 * plane;
  __bf16* p2h = ws + 2 * plane; __bf16* p2l = ws + 3 * plane;
  __bf16* z1h = ws + 4 * plane; __bf16* z1l = ws + 5 * plane;
  __bf16* z2h = ws + 6 * plane; __bf16* z2l = ws + 7 * plane;
  const int per = (HW / 128) * (HW / 128);        // 64 partials per batch
  float* part_sum = (float*)(ws + 8 * plane);     // [2][N][per]
  float* part_cnt = part_sum + (size_t)2 * N * per;

  dim3 bA(256), gA(HW / 32, N);
  normalize_split_kernel<<<gA, bA, 0, stream>>>(p1, p1h, p1l);
  normalize_split_kernel<<<gA, bA, 0, stream>>>(p2, p2h, p2l);
  normalize_split_kernel<<<gA, bA, 0, stream>>>(z1, z1h, z1l);
  normalize_split_kernel<<<gA, bA, 0, stream>>>(z2, z2h, z2l);

  dim3 bG(256), gG(HW / 128, HW / 128, N);        // 8 x 8 x N
  cossim_wmma_kernel<<<gG, bG, 0, stream>>>(p1h, p1l, z2h, z2l, c1, c2,
                                            part_sum, part_cnt);
  cossim_wmma_kernel<<<gG, bG, 0, stream>>>(p2h, p2l, z1h, z1l, c2, c1,
                                            part_sum + (size_t)N * per,
                                            part_cnt + (size_t)N * per);

  finalize_kernel<<<1, 256, 0, stream>>>(part_sum, part_cnt, 2 * N, per,
                                         (float*)d_out);
}
